// VQVAE_23416161698011
// MI455X (gfx1250) — compile-verified, matrix operand order checked by reading, not run on hardware
//
#include <hip/hip_runtime.h>
#include <stdint.h>

// ---------------------------------------------------------------------------
// VQ-VAE forward for MI455X (gfx1250, wave32).
// Convs are implicit GEMMs on the WMMA pipe with register blocking:
//   each wave computes a (16*MB) x (16*NB) output tile using MB*NB
//   v_wmma_f32_16x16x32_bf16 per 32-deep K step (A reused NB times,
//   B reused MB times).  AL=true instantiations (Cin % 32 == 0, i.e. every
//   layer except the 3-channel input convs) have a branch-free A path:
//   in-bounds -> two contiguous 16B loads, out-of-bounds (padding) -> zero
//   fragment, no scalar gathers.
// Activations: bf16 NHWC (K contiguous). Conv outputs: f32 NHWC for
// group-norm statistics. Weights repacked per call into per-lane WMMA
// B-fragment layout (contiguous 32B per lane per K-tile).
// ---------------------------------------------------------------------------

typedef __bf16 bf16;
typedef __attribute__((ext_vector_type(8)))  __bf16 bf16x8;
typedef __attribute__((ext_vector_type(16))) __bf16 bf16x16;
typedef __attribute__((ext_vector_type(8)))  float  f32x8;

#define LRELU_SLOPE 0.01f
#define GN_EPS      1e-5f
#define VQ_BETA     0.3f

__device__ __forceinline__ float lrelu_f(float v) {
    return v >= 0.f ? v : LRELU_SLOPE * v;
}

// ---------------------------------------------------------------------------
// Register-blocked implicit-GEMM conv, bf16 in/weights, f32 accumulate.
// mode: 0 = f32 NHWC out, 1 = bf16 NHWC out, 2 = f32 NCHW out + sigmoid
// Requires: Wo % (16*MB) == 0 (M tile stays within one output row),
//           Cout % (16*NB) == 0; AL requires Cin % 32 == 0.
// ---------------------------------------------------------------------------
template <int MB, int NB, bool AL>
__global__ __launch_bounds__(256) void k_conv_wmma(
    const bf16* __restrict__ in,    // NHWC, C = Cin
    const bf16* __restrict__ wpk,   // packed B fragments
    const float* __restrict__ bias, // may be null
    float* __restrict__ out_f32,
    bf16* __restrict__ out_bf,
    int B, int H, int W, int Cin, int Cinp,
    int Cout /*padded mult of 16*/, int Coutr,
    int KH, int KW, int stride, int pad,
    int Ho, int Wo, int mode, int lrelu_flag)
{
    const int lane  = threadIdx.x & 31;
    const int64_t wave = (int64_t)blockIdx.x * (blockDim.x >> 5) + (threadIdx.x >> 5);
    const int ntn  = Cout >> 4;     // 16-wide N tiles
    const int ntn2 = ntn / NB;      // blocked N tiles
    const int64_t M_total = (int64_t)B * Ho * Wo;
    const int64_t tiles = (M_total / (16 * MB)) * ntn2;
    if (wave >= tiles) return;

    const int64_t tm  = wave / ntn2;
    const int     tn2 = (int)(wave - tm * ntn2);
    const int64_t mBase = tm * (16 * MB);
    const int HoWo = Ho * Wo;
    const int b   = (int)(mBase / HoWo);
    const int rem = (int)(mBase - (int64_t)b * HoWo);
    const int oy  = rem / Wo;
    const int oxB = rem - oy * Wo;  // Wo % (16*MB) == 0 -> whole tile in one row

    const int half   = lane >> 4;
    const int mlane  = lane & 15;
    const int tnBase = tn2 * NB;    // in 16-col tile units
    const int KT     = Cinp >> 5;

    f32x8 acc[MB][NB];
#pragma unroll
    for (int mi = 0; mi < MB; ++mi)
#pragma unroll
        for (int ni = 0; ni < NB; ++ni) acc[mi][ni] = (f32x8){};

    for (int ky = 0; ky < KH; ++ky) {
        const int iy = oy * stride - pad + ky;
        const bool rowOK = (iy >= 0) && (iy < H);
        for (int kx = 0; kx < KW; ++kx) {
            const int tap = ky * KW + kx;
            bool ok[MB];
            const bf16* ap[MB];
#pragma unroll
            for (int mi = 0; mi < MB; ++mi) {
                const int ox = oxB + mi * 16 + mlane;
                const int ix = ox * stride - pad + kx;
                ok[mi] = rowOK && (ix >= 0) && (ix < W);
                ap[mi] = in + (((int64_t)b * H + iy) * W + ix) * Cin;
            }
            const bf16* wtap = wpk + ((size_t)tap * KT) * ((size_t)ntn * 512) +
                               (size_t)tnBase * 512 + (size_t)lane * 16;
#pragma unroll 2
            for (int kt = 0; kt < KT; ++kt) {
                const int k0 = kt << 5;
                // ---- A fragments (MB of them) ----
                bf16x16 a[MB];
#pragma unroll
                for (int mi = 0; mi < MB; ++mi) {
                    if (AL) {
                        if (ok[mi]) {
                            const bf16* p = ap[mi] + k0 + half * 8;
                            bf16x8 lo = *(const bf16x8*)p;
                            bf16x8 hi = *(const bf16x8*)(p + 16);
#pragma unroll
                            for (int e = 0; e < 8; ++e) {
                                a[mi][e] = lo[e]; a[mi][e + 8] = hi[e];
                            }
                        } else {
#pragma unroll
                            for (int e = 0; e < 16; ++e) a[mi][e] = (bf16)0.0f;
                        }
                    } else {
                        if (ok[mi] && (k0 + 32) <= Cin) {
                            const bf16* p = ap[mi] + k0 + half * 8;
                            bf16x8 lo = *(const bf16x8*)p;
                            bf16x8 hi = *(const bf16x8*)(p + 16);
#pragma unroll
                            for (int e = 0; e < 8; ++e) {
                                a[mi][e] = lo[e]; a[mi][e + 8] = hi[e];
                            }
                        } else {
#pragma unroll
                            for (int e = 0; e < 16; ++e) {
                                const int K  = ((e >= 8) ? 16 : 0) + half * 8 + (e & 7);
                                const int kk = k0 + K;
                                a[mi][e] = (ok[mi] && kk < Cin) ? ap[mi][kk]
                                                                : (bf16)0.0f;
                            }
                        }
                    }
                }
                // ---- B fragments (NB contiguous 16-col tiles) ----
                const bf16* bp0 = wtap + (size_t)kt * ((size_t)ntn * 512);
                if (kt + 1 < KT)
                    __builtin_prefetch((const void*)(bp0 + (size_t)ntn * 512), 0, 0);
#pragma unroll
                for (int ni = 0; ni < NB; ++ni) {
                    const bf16x16 bb = *(const bf16x16*)(bp0 + (size_t)ni * 512);
#pragma unroll
                    for (int mi = 0; mi < MB; ++mi) {
                        acc[mi][ni] = __builtin_amdgcn_wmma_f32_16x16x32_bf16(
                            false, a[mi], false, bb, (short)0, acc[mi][ni],
                            false, false);
                    }
                }
            }
        }
    }

    // D layout: VGPR r -> M = r + 8*half, N = lane%16
#pragma unroll
    for (int ni = 0; ni < NB; ++ni) {
        const int n = (tnBase + ni) * 16 + mlane;
        const float bv = (bias && n < Coutr) ? bias[n] : 0.f;
#pragma unroll
        for (int mi = 0; mi < MB; ++mi) {
#pragma unroll
            for (int r = 0; r < 8; ++r) {
                const int row = mi * 16 + (half << 3) + r;
                const int oox = oxB + row;
                float v = acc[mi][ni][r] + bv;
                if (lrelu_flag) v = lrelu_f(v);
                const size_t oNHWC = (((size_t)b * Ho + oy) * Wo + oox) * Cout + n;
                if (mode == 0) {
                    out_f32[oNHWC] = v;
                } else if (mode == 1) {
                    out_bf[oNHWC] = (bf16)v;
                } else {
                    if (n < Coutr) {
                        const float s = 1.f / (1.f + expf(-v));
                        out_f32[(((size_t)b * Coutr + n) * Ho + oy) * Wo + oox] = s;
                    }
                }
            }
        }
    }
}

// ---------------------------------------------------------------------------
// Weight repack: OIHW f32 -> per-lane WMMA B-fragment layout, bf16, zero-pad
// cin to mult-32 and cout to mult-16.
// packed[((tap*KT + kt)*NT + nt)*512 + lane*16 + e], with
//   N = nt*16 + lane%16,  K = kt*32 + ((e>=8)?16:0) + (lane>>4)*8 + (e&7)
// ---------------------------------------------------------------------------
__global__ __launch_bounds__(256) void k_pack_w(
    const float* __restrict__ w, bf16* __restrict__ wpk,
    int Cout, int Cin, int KH, int KW, int Cinp, int Coutp)
{
    const int KT = Cinp >> 5, NT = Coutp >> 4;
    const int64_t total = (int64_t)KH * KW * KT * NT * 512;
    for (int64_t p = (int64_t)blockIdx.x * blockDim.x + threadIdx.x; p < total;
         p += (int64_t)gridDim.x * blockDim.x) {
        const int64_t tile = p >> 9;
        const int e16  = (int)(p & 511);
        const int lane = e16 >> 4, j = e16 & 15;
        const int half = lane >> 4, n16 = lane & 15;
        const int K = ((j >= 8) ? 16 : 0) + half * 8 + (j & 7);
        const int nt = (int)(tile % NT);
        const int64_t t2 = tile / NT;
        const int kt = (int)(t2 % KT);
        const int tap = (int)(t2 / KT);
        const int ky = tap / KW, kx = tap % KW;
        const int cin = (kt << 5) + K;
        const int cout = (nt << 4) + n16;
        float v = 0.f;
        if (cin < Cin && cout < Cout)
            v = w[(((int64_t)cout * Cin + cin) * KH + ky) * KW + kx];
        wpk[p] = (bf16)v;
    }
}

// ---------------------------------------------------------------------------
// Pointwise / reduction kernels
// ---------------------------------------------------------------------------
__global__ __launch_bounds__(256) void k_x_to_nhwc(
    const float* __restrict__ x, bf16* __restrict__ out, int B, int C, int H, int W)
{
    const int64_t total = (int64_t)B * C * H * W;
    for (int64_t i = (int64_t)blockIdx.x * blockDim.x + threadIdx.x; i < total;
         i += (int64_t)gridDim.x * blockDim.x) {
        const int c = (int)(i % C);
        int64_t t = i / C;
        const int w = (int)(t % W); t /= W;
        const int h = (int)(t % H);
        const int b = (int)(t / H);
        out[i] = (bf16)x[(((int64_t)b * C + c) * H + h) * W + w];
    }
}

__global__ __launch_bounds__(256) void k_f32_to_bf16(
    const float* __restrict__ in, bf16* __restrict__ out, int64_t n)
{
    for (int64_t i = (int64_t)blockIdx.x * blockDim.x + threadIdx.x; i < n;
         i += (int64_t)gridDim.x * blockDim.x)
        out[i] = (bf16)in[i];
}

// GroupNorm over f32 NHWC conv output; optional post-affine lrelu and bf16
// residual add; writes bf16 NHWC. One block per (batch, group).
__global__ __launch_bounds__(256) void k_groupnorm(
    const float* __restrict__ xin, const float* __restrict__ gamma,
    const float* __restrict__ beta, const bf16* __restrict__ residual,
    bf16* __restrict__ out, int B, int C, int H, int W, int groups, int post_lrelu)
{
    const int bg = blockIdx.x;
    const int b = bg / groups, g = bg % groups;
    const int cpg = C / groups;
    const int64_t HW = (int64_t)H * W;
    const int64_t cnt = HW * cpg;

    float s = 0.f, s2 = 0.f;
    for (int64_t i = threadIdx.x; i < cnt; i += blockDim.x) {
        const int64_t hw = i / cpg;
        const int cc = (int)(i % cpg);
        const float v = xin[((int64_t)b * HW + hw) * C + g * cpg + cc];
        s += v; s2 += v * v;
    }
    __shared__ float sh[512];
    sh[threadIdx.x] = s; sh[256 + threadIdx.x] = s2;
    __syncthreads();
    for (int st = 128; st > 0; st >>= 1) {
        if ((int)threadIdx.x < st) {
            sh[threadIdx.x] += sh[threadIdx.x + st];
            sh[256 + threadIdx.x] += sh[256 + threadIdx.x + st];
        }
        __syncthreads();
    }
    const float mean = sh[0] / (float)cnt;
    const float var  = sh[256] / (float)cnt - mean * mean;
    const float inv  = rsqrtf(var + GN_EPS);

    for (int64_t i = threadIdx.x; i < cnt; i += blockDim.x) {
        const int64_t hw = i / cpg;
        const int cc = (int)(i % cpg);
        const int c = g * cpg + cc;
        const int64_t idx = ((int64_t)b * HW + hw) * C + c;
        float v = (xin[idx] - mean) * inv * gamma[c] + beta[c];
        if (post_lrelu) v = lrelu_f(v);
        if (residual) v += (float)residual[idx];
        out[idx] = (bf16)v;
    }
}

__global__ __launch_bounds__(256) void k_upsample2x(
    const bf16* __restrict__ in, bf16* __restrict__ out, int B, int C, int H, int W)
{
    const int W2 = 2 * W, H2 = 2 * H;
    const int64_t total = (int64_t)B * H2 * W2 * C;
    for (int64_t i = (int64_t)blockIdx.x * blockDim.x + threadIdx.x; i < total;
         i += (int64_t)gridDim.x * blockDim.x) {
        const int c = (int)(i % C);
        int64_t t = i / C;
        const int ox = (int)(t % W2); t /= W2;
        const int oy = (int)(t % H2);
        const int b = (int)(t / H2);
        out[i] = in[(((int64_t)b * H + (oy >> 1)) * W + (ox >> 1)) * C + c];
    }
}

__global__ __launch_bounds__(256) void k_cbnorm(
    const float* __restrict__ cb, float* __restrict__ cbn, int K, int D)
{
    const int k = blockIdx.x;
    float s = 0.f;
    for (int d = threadIdx.x; d < D; d += blockDim.x) {
        const float v = cb[(int64_t)k * D + d];
        s += v * v;
    }
    __shared__ float sh[256];
    sh[threadIdx.x] = s; __syncthreads();
    for (int st = 128; st > 0; st >>= 1) {
        if ((int)threadIdx.x < st) sh[threadIdx.x] += sh[threadIdx.x + st];
        __syncthreads();
    }
    if (threadIdx.x == 0) cbn[k] = sh[0];
}

__global__ __launch_bounds__(256) void k_argmin(
    const float* __restrict__ dots, const float* __restrict__ cbn,
    int N, int K, int* __restrict__ idx)
{
    const int n = blockIdx.x * blockDim.x + threadIdx.x;
    if (n >= N) return;
    float best = 3.402823466e38f;
    int bi = 0;
    for (int k = 0; k < K; ++k) {
        const float d = cbn[k] - 2.f * dots[(int64_t)n * K + k];
        if (d < best) { best = d; bi = k; }
    }
    idx[n] = bi;
}

// Gather zq (bf16 for decoder) + per-block partial sums of (z - zq)^2.
__global__ __launch_bounds__(256) void k_zq_loss(
    const float* __restrict__ z, const float* __restrict__ cb,
    const int* __restrict__ idx, bf16* __restrict__ zqbf,
    float* __restrict__ partial, int N, int D)
{
    const int64_t total = (int64_t)N * D;
    float acc = 0.f;
    for (int64_t i = (int64_t)blockIdx.x * blockDim.x + threadIdx.x; i < total;
         i += (int64_t)gridDim.x * blockDim.x) {
        const int n = (int)(i / D);
        const int d = (int)(i % D);
        const float zq = cb[(int64_t)idx[n] * D + d];
        const float diff = z[i] - zq;
        acc += diff * diff;
        zqbf[i] = (bf16)zq;
    }
    __shared__ float sh[256];
    sh[threadIdx.x] = acc; __syncthreads();
    for (int st = 128; st > 0; st >>= 1) {
        if ((int)threadIdx.x < st) sh[threadIdx.x] += sh[threadIdx.x + st];
        __syncthreads();
    }
    if (threadIdx.x == 0) partial[blockIdx.x] = sh[0];
}

__global__ __launch_bounds__(256) void k_reduce_loss(
    const float* __restrict__ partial, int n, float scale, float* __restrict__ out)
{
    float s = 0.f;
    for (int i = threadIdx.x; i < n; i += blockDim.x) s += partial[i];
    __shared__ float sh[256];
    sh[threadIdx.x] = s; __syncthreads();
    for (int st = 128; st > 0; st >>= 1) {
        if ((int)threadIdx.x < st) sh[threadIdx.x] += sh[threadIdx.x + st];
        __syncthreads();
    }
    if (threadIdx.x == 0) out[0] = sh[0] * scale;
}

// ---------------------------------------------------------------------------
// Orchestration
// ---------------------------------------------------------------------------
extern "C" void kernel_launch(void* const* d_in, const int* in_sizes, int n_in,
                              void* d_out, int out_size, void* d_ws, size_t ws_size,
                              hipStream_t stream)
{
    (void)in_sizes; (void)n_in; (void)ws_size;
    char* ws = (char*)d_ws;
    float* doutf = (float*)d_out;

    // -------- workspace layout --------
    size_t off = 0;
    auto alloc = [&](size_t bytes) -> size_t {
        size_t o = off;
        off = (off + bytes + 255) & ~(size_t)255;
        return o;
    };
    const size_t EL_ACT = (size_t)8 * 256 * 256 * 128;   // max activation elems
    const size_t ACTA = alloc(EL_ACT * 2);
    const size_t ACTB = alloc(EL_ACT * 2);
    const size_t ACTR = alloc(EL_ACT * 2);
    const size_t ACTF = alloc(EL_ACT * 4);
    const size_t WPK  = alloc(20u * 1024 * 1024);
    const size_t Zo   = alloc((size_t)8192 * 256 * 4);
    const size_t ZBFo = alloc((size_t)8192 * 256 * 2);
    const size_t ZQo  = alloc((size_t)8192 * 256 * 2);
    const size_t DOTo = alloc((size_t)8192 * 512 * 4);
    const size_t CBNo = alloc(512 * 4);
    const size_t IDXo = alloc(8192 * 4);
    const size_t PARTo = alloc(1024 * 4);

    bf16* actA = (bf16*)(ws + ACTA);
    bf16* actB = (bf16*)(ws + ACTB);
    bf16* actR = (bf16*)(ws + ACTR);
    float* actF = (float*)(ws + ACTF);
    float* Zf = (float*)(ws + Zo);
    bf16* Zbf = (bf16*)(ws + ZBFo);
    bf16* ZQbf = (bf16*)(ws + ZQo);
    float* DOTS = (float*)(ws + DOTo);
    float* CBN = (float*)(ws + CBNo);
    int* IDX = (int*)(ws + IDXo);
    float* PART = (float*)(ws + PARTo);

    // -------- launch helpers --------
    size_t woff = WPK;
    auto packW = [&](const void* w, int Cout, int Cin, int KH, int KW) -> const bf16* {
        const int Cinp = (Cin + 31) & ~31, Coutp = (Cout + 15) & ~15;
        const size_t elems = (size_t)KH * KW * (Cinp / 32) * (Coutp / 16) * 512;
        bf16* dst = (bf16*)(ws + woff);
        woff += (elems * 2 + 255) & ~(size_t)255;
        int blocks = (int)((elems + 255) / 256);
        if (blocks > 4096) blocks = 4096;
        k_pack_w<<<blocks, 256, 0, stream>>>((const float*)w, dst, Cout, Cin, KH, KW,
                                             Cinp, Coutp);
        return dst;
    };
    auto conv = [&](const bf16* in, const bf16* wpk, const float* bias,
                    float* of32, bf16* obf,
                    int B, int H, int W, int Cin, int Cout, int Coutr,
                    int KH, int KW, int stride, int pad, int mode, int lre) {
        const int Cinp = (Cin + 31) & ~31;
        const int Ho = (H + 2 * pad - KH) / stride + 1;
        const int Wo = (W + 2 * pad - KW) / stride + 1;
        const int64_t M_total = (int64_t)B * Ho * Wo;
        if ((Cout % 64) == 0) {
            const int64_t tiles = (M_total / 32) * (Cout / 64);
            const int blocks = (int)((tiles * 32 + 255) / 256);
            if ((Cin % 32) == 0) {
                k_conv_wmma<2, 4, true><<<blocks, 256, 0, stream>>>(
                    in, wpk, bias, of32, obf, B, H, W, Cin, Cinp, Cout, Coutr,
                    KH, KW, stride, pad, Ho, Wo, mode, lre);
            } else {
                k_conv_wmma<2, 4, false><<<blocks, 256, 0, stream>>>(
                    in, wpk, bias, of32, obf, B, H, W, Cin, Cinp, Cout, Coutr,
                    KH, KW, stride, pad, Ho, Wo, mode, lre);
            }
        } else {
            const int64_t tiles = (M_total / 32) * (Cout / 16);
            const int blocks = (int)((tiles * 32 + 255) / 256);
            k_conv_wmma<2, 1, true><<<blocks, 256, 0, stream>>>(
                in, wpk, bias, of32, obf, B, H, W, Cin, Cinp, Cout, Coutr,
                KH, KW, stride, pad, Ho, Wo, mode, lre);
        }
    };
    auto gnorm = [&](const float* xin, const float* g, const float* b,
                     const bf16* res, bf16* out, int B, int C, int H, int W, int post) {
        k_groupnorm<<<B * 32, 256, 0, stream>>>(xin, g, b, res, out, B, C, H, W, 32, post);
    };
    auto upsample = [&](const bf16* in, bf16* out, int B, int C, int H, int W) {
        const int64_t total = (int64_t)B * C * 4 * H * W;
        int blocks = (int)((total + 255) / 256);
        if (blocks > 16384) blocks = 16384;
        k_upsample2x<<<blocks, 256, 0, stream>>>(in, out, B, C, H, W);
    };

    // -------- input pointers (setup_inputs dict order, depth-first) --------
    int ii = 0;
    const float* X = (const float*)d_in[ii++];                       // (8,3,256,256)
    const float *e0cw = (const float*)d_in[ii++], *e0cb = (const float*)d_in[ii++],
                *e0g = (const float*)d_in[ii++], *e0b = (const float*)d_in[ii++],
                *e0pw = (const float*)d_in[ii++], *e0pb = (const float*)d_in[ii++];
    const float *e1cw = (const float*)d_in[ii++], *e1cb = (const float*)d_in[ii++],
                *e1g = (const float*)d_in[ii++], *e1b = (const float*)d_in[ii++];
    const float *e2cw = (const float*)d_in[ii++], *e2cb = (const float*)d_in[ii++],
                *e2g = (const float*)d_in[ii++], *e2b = (const float*)d_in[ii++];
    const float *e3cw = (const float*)d_in[ii++], *e3cb = (const float*)d_in[ii++],
                *e3g = (const float*)d_in[ii++], *e3b = (const float*)d_in[ii++];
    const float *e4cw = (const float*)d_in[ii++], *e4cb = (const float*)d_in[ii++],
                *e4g = (const float*)d_in[ii++], *e4b = (const float*)d_in[ii++];
    const float *e5cw = (const float*)d_in[ii++], *e5cb = (const float*)d_in[ii++],
                *e5g = (const float*)d_in[ii++], *e5b = (const float*)d_in[ii++];
    const float *efw = (const float*)d_in[ii++], *efb = (const float*)d_in[ii++];
    const float *CB = (const float*)d_in[ii++];                      // (512,256)
    const float *d0cw = (const float*)d_in[ii++], *d0cb = (const float*)d_in[ii++],
                *d0g = (const float*)d_in[ii++], *d0b = (const float*)d_in[ii++];
    const float *d1cw = (const float*)d_in[ii++], *d1cb = (const float*)d_in[ii++],
                *d1g = (const float*)d_in[ii++], *d1b = (const float*)d_in[ii++],
                *d1pw = (const float*)d_in[ii++], *d1pb = (const float*)d_in[ii++];
    const float *d2cw = (const float*)d_in[ii++], *d2cb = (const float*)d_in[ii++],
                *d2g = (const float*)d_in[ii++], *d2b = (const float*)d_in[ii++];
    const float *d3cw = (const float*)d_in[ii++], *d3cb = (const float*)d_in[ii++],
                *d3g = (const float*)d_in[ii++], *d3b = (const float*)d_in[ii++],
                *d3pw = (const float*)d_in[ii++], *d3pb = (const float*)d_in[ii++];
    const float *d4cw = (const float*)d_in[ii++], *d4cb = (const float*)d_in[ii++],
                *d4g = (const float*)d_in[ii++], *d4b = (const float*)d_in[ii++];
    const float *d5cw = (const float*)d_in[ii++], *d5cb = (const float*)d_in[ii++],
                *d5g = (const float*)d_in[ii++], *d5b = (const float*)d_in[ii++];
    const float *dfw = (const float*)d_in[ii++], *dfb = (const float*)d_in[ii++];

    // -------- pack all weights (inputs may change per call) --------
    const bf16* We0c = packW(e0cw, 128, 3, 3, 3);
    const bf16* We0p = packW(e0pw, 128, 3, 1, 1);
    const bf16* We1c = packW(e1cw, 128, 128, 4, 4);
    const bf16* We2c = packW(e2cw, 128, 128, 3, 3);
    const bf16* We3c = packW(e3cw, 256, 128, 4, 4);
    const bf16* We4c = packW(e4cw, 256, 256, 3, 3);
    const bf16* We5c = packW(e5cw, 256, 256, 4, 4);
    const bf16* Wef  = packW(efw, 256, 256, 1, 1);
    const bf16* Wcb  = packW(CB, 512, 256, 1, 1);    // codebook as 1x1 conv
    const bf16* Wd0c = packW(d0cw, 256, 256, 3, 3);
    const bf16* Wd1c = packW(d1cw, 128, 256, 3, 3);
    const bf16* Wd1p = packW(d1pw, 128, 256, 1, 1);
    const bf16* Wd2c = packW(d2cw, 128, 128, 3, 3);
    const bf16* Wd3c = packW(d3cw, 64, 128, 3, 3);
    const bf16* Wd3p = packW(d3pw, 64, 128, 1, 1);
    const bf16* Wd4c = packW(d4cw, 64, 64, 3, 3);
    const bf16* Wd5c = packW(d5cw, 64, 64, 3, 3);
    const bf16* Wdf  = packW(dfw, 3, 64, 1, 1);      // Coutp = 16 (padded)

    // -------- input conversion NCHW f32 -> NHWC bf16 --------
    k_x_to_nhwc<<<6144, 256, 0, stream>>>(X, actA, 8, 3, 256, 256);

    // -------- encoder --------
    // res(3->128) @256
    conv(actA, We0c, e0cb, actF, nullptr, 8, 256, 256, 3, 128, 128, 3, 3, 1, 1, 0, 0);
    conv(actA, We0p, e0pb, nullptr, actR, 8, 256, 256, 3, 128, 128, 1, 1, 1, 0, 1, 0);
    gnorm(actF, e0g, e0b, actR, actB, 8, 128, 256, 256, 1);
    // down(128->128) -> 128 (conv -> lrelu -> gn)
    conv(actB, We1c, e1cb, actF, nullptr, 8, 256, 256, 128, 128, 128, 4, 4, 2, 1, 0, 1);
    gnorm(actF, e1g, e1b, nullptr, actA, 8, 128, 128, 128, 0);
    // res(128) @128
    conv(actA, We2c, e2cb, actF, nullptr, 8, 128, 128, 128, 128, 128, 3, 3, 1, 1, 0, 0);
    gnorm(actF, e2g, e2b, actA, actB, 8, 128, 128, 128, 1);
    // down(128->256) -> 64
    conv(actB, We3c, e3cb, actF, nullptr, 8, 128, 128, 128, 256, 256, 4, 4, 2, 1, 0, 1);
    gnorm(actF, e3g, e3b, nullptr, actA, 8, 256, 64, 64, 0);
    // res(256) @64
    conv(actA, We4c, e4cb, actF, nullptr, 8, 64, 64, 256, 256, 256, 3, 3, 1, 1, 0, 0);
    gnorm(actF, e4g, e4b, actA, actB, 8, 256, 64, 64, 1);
    // down(256->256) -> 32
    conv(actB, We5c, e5cb, actF, nullptr, 8, 64, 64, 256, 256, 256, 4, 4, 2, 1, 0, 1);
    gnorm(actF, e5g, e5b, nullptr, actA, 8, 256, 32, 32, 0);
    // enc_final 1x1 + lrelu -> z (f32 NHWC)
    conv(actA, Wef, efb, Zf, nullptr, 8, 32, 32, 256, 256, 256, 1, 1, 1, 0, 0, 1);
    k_f32_to_bf16<<<8192, 256, 0, stream>>>(Zf, Zbf, (int64_t)8192 * 256);

    // -------- vector quantize --------
    // dots = z @ cb^T via WMMA (1x1 "conv": M=8192, K=256, N=512)
    conv(Zbf, Wcb, nullptr, DOTS, nullptr, 8, 32, 32, 256, 512, 512, 1, 1, 1, 0, 0, 0);
    k_cbnorm<<<512, 256, 0, stream>>>(CB, CBN, 512, 256);
    k_argmin<<<32, 256, 0, stream>>>(DOTS, CBN, 8192, 512, IDX);
    // forward loss value = (1+beta)*mse(z,zq); zq (bf16) feeds decoder
    k_zq_loss<<<1024, 256, 0, stream>>>(Zf, CB, IDX, ZQbf, PART, 8192, 256);
    k_reduce_loss<<<1, 256, 0, stream>>>(PART, 1024,
                                         (1.0f + VQ_BETA) / (8192.0f * 256.0f),
                                         doutf + (out_size - 1));

    // -------- decoder --------
    // res(256) @32
    conv(ZQbf, Wd0c, d0cb, actF, nullptr, 8, 32, 32, 256, 256, 256, 3, 3, 1, 1, 0, 0);
    gnorm(actF, d0g, d0b, ZQbf, actA, 8, 256, 32, 32, 1);
    upsample(actA, actB, 8, 256, 32, 32);                 // -> 64
    // res(256->128) @64
    conv(actB, Wd1c, d1cb, actF, nullptr, 8, 64, 64, 256, 128, 128, 3, 3, 1, 1, 0, 0);
    conv(actB, Wd1p, d1pb, nullptr, actR, 8, 64, 64, 256, 128, 128, 1, 1, 1, 0, 1, 0);
    gnorm(actF, d1g, d1b, actR, actA, 8, 128, 64, 64, 1);
    // res(128) @64
    conv(actA, Wd2c, d2cb, actF, nullptr, 8, 64, 64, 128, 128, 128, 3, 3, 1, 1, 0, 0);
    gnorm(actF, d2g, d2b, actA, actB, 8, 128, 64, 64, 1);
    upsample(actB, actA, 8, 128, 64, 64);                 // -> 128
    // res(128->64) @128
    conv(actA, Wd3c, d3cb, actF, nullptr, 8, 128, 128, 128, 64, 64, 3, 3, 1, 1, 0, 0);
    conv(actA, Wd3p, d3pb, nullptr, actR, 8, 128, 128, 128, 64, 64, 1, 1, 1, 0, 1, 0);
    gnorm(actF, d3g, d3b, actR, actB, 8, 64, 128, 128, 1);
    // res(64) @128
    conv(actB, Wd4c, d4cb, actF, nullptr, 8, 128, 128, 64, 64, 64, 3, 3, 1, 1, 0, 0);
    gnorm(actF, d4g, d4b, actB, actA, 8, 64, 128, 128, 1);
    upsample(actA, actB, 8, 64, 128, 128);                // -> 256
    // res(64) @256
    conv(actB, Wd5c, d5cb, actF, nullptr, 8, 256, 256, 64, 64, 64, 3, 3, 1, 1, 0, 0);
    gnorm(actF, d5g, d5b, actB, actA, 8, 64, 256, 256, 1);
    // final 1x1 (64->3, padded to 16) + sigmoid, writes recon f32 NCHW to d_out
    conv(actA, Wdf, dfb, doutf, nullptr, 8, 256, 256, 64, 16, 3, 1, 1, 1, 0, 2, 0);
}
